// TestModule_76450417868925
// MI455X (gfx1250) — matrix operand order repaired and checked by reading, compiled, not run on hardware
//
#include <hip/hip_runtime.h>

// ---------------------------------------------------------------------------
// Varlen non-causal flash attention for MI455X (gfx1250, wave32).
//  * QK^T and PV on v_wmma_f32_16x16x32_bf16 (f32 accumulate)
//  * K/V tiles DMA'd global->LDS by the Tensor Data Mover (double buffered,
//    synced with s_wait_tensorcnt), overlapping the WMMA compute
//  * single cooperative f32->bf16 conversion pass per tile (K row-major,
//    V transposed), so waves read bf16 fragments with no redundant converts
//  * softmax row reductions via DPP16 butterflies (no ds_bpermute)
// One workgroup (4 wave32) = 64 q rows of one (seq, head); 32-key tiles.
// ---------------------------------------------------------------------------

#define H_HEADS 8
#define HDIM    128
#define SMAX    2048
#define QTILE_BLOCK 64
#define QTILE_WAVE  16
#define KTILE       32

typedef __attribute__((ext_vector_type(16))) __bf16 v16bf;
typedef __attribute__((ext_vector_type(8)))  __bf16 v8bf;
typedef __attribute__((ext_vector_type(8)))  float  v8f;
typedef __attribute__((ext_vector_type(4)))  float  v4f;

#define HAVE_TDM __has_builtin(__builtin_amdgcn_tensor_load_to_lds)
#define HAVE_DPP __has_builtin(__builtin_amdgcn_mov_dpp)

// ---- 16-lane butterfly reductions (DPP16: pure VALU, no LDS traffic) ------
#if HAVE_DPP
template <int CTRL>
__device__ __forceinline__ float dpp_permf(float x) {
  return __int_as_float(
      __builtin_amdgcn_mov_dpp(__float_as_int(x), CTRL, 0xF, 0xF, true));
}
__device__ __forceinline__ float red16_max(float x) {
  x = fmaxf(x, dpp_permf<0x0B1>(x));  // quad_perm(1,0,3,2)  : xor 1
  x = fmaxf(x, dpp_permf<0x04E>(x));  // quad_perm(2,3,0,1)  : xor 2
  x = fmaxf(x, dpp_permf<0x141>(x));  // row_half_mirror     : join quads
  x = fmaxf(x, dpp_permf<0x140>(x));  // row_mirror          : join octs
  return x;
}
__device__ __forceinline__ float red16_sum(float x) {
  x += dpp_permf<0x0B1>(x);
  x += dpp_permf<0x04E>(x);
  x += dpp_permf<0x141>(x);
  x += dpp_permf<0x140>(x);
  return x;
}
#else
__device__ __forceinline__ float red16_max(float x) {
#pragma unroll
  for (int m = 8; m >= 1; m >>= 1) x = fmaxf(x, __shfl_xor(x, m, 32));
  return x;
}
__device__ __forceinline__ float red16_sum(float x) {
#pragma unroll
  for (int m = 8; m >= 1; m >>= 1) x += __shfl_xor(x, m, 32);
  return x;
}
#endif

__device__ __forceinline__ void wave_lds_fence() {
#if __has_builtin(__builtin_amdgcn_s_wait_dscnt)
  __builtin_amdgcn_s_wait_dscnt(0);
  asm volatile("" ::: "memory");
#else
  asm volatile("s_wait_dscnt 0" ::: "memory");
#endif
}

// ---- Tensor Data Mover: 2D f32 tile (KTILE rows x HDIM cols) -> LDS -------
#if HAVE_TDM
typedef unsigned int u32x4 __attribute__((ext_vector_type(4)));
typedef int          i32x4 __attribute__((ext_vector_type(4)));
typedef int          i32x8 __attribute__((ext_vector_type(8)));

__device__ __forceinline__ void tdm_load_tile(unsigned lds_addr,
                                              const float* gptr, int rows) {
  unsigned long long ga = (unsigned long long)(uintptr_t)gptr;
  // D# group0: count=1 | lds_addr | global_addr[56:0] | type=2 (bits 127:126)
  u32x4 g0 = { 1u, lds_addr, (unsigned)ga,
               (unsigned)(ga >> 32) | 0x80000000u };
  // D# group1: data_size=4B, tensor_dim0=HDIM, tensor_dim1=rows (zero-fill
  // past ragged tail), tile = HDIM x KTILE, dim0 stride = H*D elements.
  i32x8 g1 = { (int)(2u << 16),
               (int)((unsigned)HDIM << 16),
               (int)((unsigned)rows << 16),
               (int)((unsigned)HDIM << 16),
               (int)KTILE,
               (int)(H_HEADS * HDIM),
               0, 0 };
  i32x4 z4 = { 0, 0, 0, 0 };
#if __clang_major__ >= 23
  i32x8 z8 = { 0, 0, 0, 0, 0, 0, 0, 0 };
  __builtin_amdgcn_tensor_load_to_lds(g0, g1, z4, z4, z8, 0);
#else
  __builtin_amdgcn_tensor_load_to_lds(g0, g1, z4, z4, 0);
#endif
}
#endif

__global__ __launch_bounds__(128, 1)
void fa_varlen_wmma_kernel(const float* __restrict__ q,
                           const float* __restrict__ k,
                           const float* __restrict__ v,
                           const int* __restrict__ cu_q,
                           const int* __restrict__ cu_k,
                           float* __restrict__ out)
{
  __shared__ float  stK[2][KTILE * HDIM];          // 32 KB f32 K staging (TDM)
  __shared__ float  stV[2][KTILE * HDIM];          // 32 KB f32 V staging (TDM)
  __shared__ __bf16 ldsKb[KTILE * HDIM];           //  8 KB K bf16 row-major
  __shared__ __bf16 ldsvT[HDIM * KTILE];           //  8 KB V^T bf16
  __shared__ __bf16 ldsP[4][QTILE_WAVE * KTILE];   //  4 KB per-wave P staging

  const int nqt = SMAX / QTILE_BLOCK;
  const int bid = blockIdx.x;
  const int qt  = bid % nqt;
  const int h   = (bid / nqt) % H_HEADS;
  const int b   = bid / (nqt * H_HEADS);

  const int qstart = cu_q[b];
  const int qlen   = cu_q[b + 1] - qstart;
  const int kstart = cu_k[b];
  const int klen   = cu_k[b + 1] - kstart;
  const int q0     = qt * QTILE_BLOCK;
  if (q0 >= qlen || klen <= 0) return;             // block-uniform exit

  const int tid   = threadIdx.x;
  const int wave  = tid >> 5;
  const int lane  = tid & 31;
  const int half  = lane >> 4;
  const int l16   = lane & 15;
  const int qrow0 = q0 + wave * QTILE_WAVE;

  const size_t rowstride = (size_t)H_HEADS * HDIM;
  const float  scale     = 0.08838834764831845f;   // 1/sqrt(128)

  const float* kbase = k + (size_t)kstart * rowstride + (size_t)h * HDIM;
  const float* vbase = v + (size_t)kstart * rowstride + (size_t)h * HDIM;

  // ---- Q A-fragments (contiguous K layout: half 0 -> K {0..15}) ----
  v16bf qa[4];
  {
    int qr = qrow0 + l16;
    if (qr >= qlen) qr = qlen - 1;
    const float* qrow = q + (size_t)(qstart + qr) * rowstride + (size_t)h * HDIM;
#pragma unroll
    for (int dc = 0; dc < 4; ++dc) {
#pragma unroll
      for (int j = 0; j < 16; ++j) {
        int kk = half * 16 + j;
        qa[dc][j] = (__bf16)qrow[dc * 32 + kk];
      }
    }
  }

  v8f o[8];
#pragma unroll
  for (int i = 0; i < 8; ++i) o[i] = (v8f){};
  float mrow[8], lrow[8];
#pragma unroll
  for (int r = 0; r < 8; ++r) { mrow[r] = -3.0e38f; lrow[r] = 0.0f; }

#if HAVE_TDM
  // prologue: DMA tile 0 into buffer 0 (overlaps Q fragment setup above)
  if (wave == 0) {
    int rem = klen > 65535 ? 65535 : klen;
    tdm_load_tile((unsigned)(uintptr_t)&stK[0][0], kbase, rem);
    tdm_load_tile((unsigned)(uintptr_t)&stV[0][0], vbase, rem);
  }
#endif

  int it = 0;
  for (int kb = 0; kb < klen; kb += KTILE, ++it) {
    const int cur = it & 1;
    __syncthreads();                 // all waves done with both buffers

#if HAVE_TDM
    if (wave == 0) {
      const int kbn = kb + KTILE;
      if (kbn < klen) {              // DMA next tile, overlaps this compute
        int rem = klen - kbn; if (rem > 65535) rem = 65535;
        tdm_load_tile((unsigned)(uintptr_t)&stK[cur ^ 1][0],
                      kbase + (size_t)kbn * rowstride, rem);
        tdm_load_tile((unsigned)(uintptr_t)&stV[cur ^ 1][0],
                      vbase + (size_t)kbn * rowstride, rem);
        __builtin_amdgcn_s_wait_tensorcnt((short)2);  // current tile landed
      } else {
        __builtin_amdgcn_s_wait_tensorcnt((short)0);
      }
    }
#else
    {  // fallback: cooperative global->LDS copy of current tile (f32)
      int krow = tid >> 2, seg = tid & 3;
      int src = kb + krow; if (src >= klen) src = klen - 1;
      const float* kp = kbase + (size_t)src * rowstride + seg * 32;
      const float* vp = vbase + (size_t)src * rowstride + seg * 32;
      float* dk = &stK[cur][krow * HDIM + seg * 32];
      float* dv = &stV[cur][krow * HDIM + seg * 32];
#pragma unroll
      for (int j = 0; j < 8; ++j) {
        ((v4f*)dk)[j] = ((const v4f*)kp)[j];
        ((v4f*)dv)[j] = ((const v4f*)vp)[j];
      }
    }
#endif
    __syncthreads();                 // stK/stV[cur] valid for all waves

    // ---- single cooperative conversion pass ----
    {
      // K: f32 -> bf16 row-major (vectorized b128 stores, cvt_pk pairs)
      int krow = tid >> 2, seg = tid & 3;
      const float* srck = &stK[cur][krow * HDIM + seg * 32];
      __bf16* dstk = &ldsKb[krow * HDIM + seg * 32];
#pragma unroll
      for (int c = 0; c < 4; ++c) {
        v8bf pk;
#pragma unroll
        for (int j = 0; j < 8; ++j) pk[j] = (__bf16)srck[c * 8 + j];
        *(v8bf*)&dstk[c * 8] = pk;
      }
      // V: f32 -> bf16 transposed (two keys packed per b32 store)
      int pairi = tid & 15, dseg = tid >> 4;
      int d0 = dseg * 16;
      const float* r0 = &stV[cur][(2 * pairi) * HDIM + d0];
      const float* r1 = &stV[cur][(2 * pairi + 1) * HDIM + d0];
      unsigned* dstv = (unsigned*)&ldsvT[0];
#pragma unroll
      for (int j = 0; j < 16; ++j) {
        unsigned short ua = __builtin_bit_cast(unsigned short, (__bf16)r0[j]);
        unsigned short ub = __builtin_bit_cast(unsigned short, (__bf16)r1[j]);
        dstv[((d0 + j) * KTILE + 2 * pairi) >> 1] =
            (unsigned)ua | ((unsigned)ub << 16);
      }
    }
    __syncthreads();                 // ldsKb / ldsvT ready

    // ---- S = Q K^T : bf16 B-fragments straight from LDS ----
    v8f s0 = (v8f){}, s1 = (v8f){};
#pragma unroll
    for (int dc = 0; dc < 4; ++dc) {
      const int dbase = dc * 32 + half * 16;
      v16bf b0 = *(const v16bf*)&ldsKb[(l16     ) * HDIM + dbase];
      v16bf b1 = *(const v16bf*)&ldsKb[(16 + l16) * HDIM + dbase];
      s0 = __builtin_amdgcn_wmma_f32_16x16x32_bf16(false, qa[dc], false, b0,
                                                   (short)0, s0, false, false);
      s1 = __builtin_amdgcn_wmma_f32_16x16x32_bf16(false, qa[dc], false, b1,
                                                   (short)0, s1, false, false);
    }

    // ---- online softmax (C layout: lane col = l16, row = half*8 + r) ----
    const float mask0 = ((kb + l16)      < klen) ? 0.0f : -1.0e30f;
    const float mask1 = ((kb + 16 + l16) < klen) ? 0.0f : -1.0e30f;
    float corr[8];
#pragma unroll
    for (int r = 0; r < 8; ++r) {
      float a0 = s0[r] * scale + mask0;
      float a1 = s1[r] * scale + mask1;
      float rm   = red16_max(fmaxf(a0, a1));
      float mnew = fmaxf(mrow[r], rm);
      float c    = __expf(mrow[r] - mnew);
      float p0   = __expf(a0 - mnew);
      float p1   = __expf(a1 - mnew);
      lrow[r] = lrow[r] * c + red16_sum(p0 + p1);
      mrow[r] = mnew;
      corr[r] = c;
      s0[r] = p0; s1[r] = p1;
    }
#pragma unroll
    for (int i = 0; i < 8; ++i) {
#pragma unroll
      for (int r = 0; r < 8; ++r) o[i][r] *= corr[r];
    }

    // ---- P: C layout -> A layout via per-wave LDS staging ----
    __bf16* ps = ldsP[wave];
#pragma unroll
    for (int r = 0; r < 8; ++r) {
      int row = half * 8 + r;
      ps[row * KTILE + l16]      = (__bf16)s0[r];
      ps[row * KTILE + 16 + l16] = (__bf16)s1[r];
    }
    wave_lds_fence();                // DS ops in-order within wave

    v8bf plo = *(const v8bf*)&ps[l16 * KTILE + half * 16];
    v8bf phi = *(const v8bf*)&ps[l16 * KTILE + half * 16 + 8];
    v16bf pa = __builtin_shufflevector(plo, phi,
                  0, 1, 2, 3, 4, 5, 6, 7, 8, 9, 10, 11, 12, 13, 14, 15);

    // ---- O += P V : one WMMA per 16-wide d-chunk (K = 32 keys) ----
#pragma unroll
    for (int dcn = 0; dcn < 8; ++dcn) {
      v16bf bv = *(const v16bf*)&ldsvT[(dcn * 16 + l16) * KTILE + half * 16];
      o[dcn] = __builtin_amdgcn_wmma_f32_16x16x32_bf16(false, pa, false, bv,
                                                       (short)0, o[dcn], false, false);
    }
  }

  // ---- finalize: O /= l, scatter to packed [T,H,D] ----
#pragma unroll
  for (int r = 0; r < 8; ++r) {
    int row = qrow0 + half * 8 + r;
    if (row < qlen) {
      float inv = 1.0f / lrow[r];
      float* orow = out + (size_t)(qstart + row) * rowstride + (size_t)h * HDIM;
#pragma unroll
      for (int dcn = 0; dcn < 8; ++dcn)
        orow[dcn * 16 + l16] = o[dcn][r] * inv;
    }
  }
}

extern "C" void kernel_launch(void* const* d_in, const int* in_sizes, int n_in,
                              void* d_out, int out_size, void* d_ws, size_t ws_size,
                              hipStream_t stream) {
  (void)n_in; (void)out_size; (void)d_ws; (void)ws_size;
  const float* q   = (const float*)d_in[0];
  const float* k   = (const float*)d_in[1];
  const float* v   = (const float*)d_in[2];
  const int*   cuq = (const int*)d_in[3];
  const int*   cuk = (const int*)d_in[4];
  float*       out = (float*)d_out;

  const int B = in_sizes[3] - 1;
  dim3 grid(B * H_HEADS * (SMAX / QTILE_BLOCK));
  dim3 block(128);
  fa_varlen_wmma_kernel<<<grid, block, 0, stream>>>(q, k, v, cuq, cuk, out);
}
